// QuantLinearORT_56152402428651
// MI455X (gfx1250) — matrix operand order repaired and checked by reading, not compiled, not run on hardware
//
#include <hip/hip_runtime.h>
#include <hip/hip_bf16.h>
#include <stdint.h>

// ---------------------------------------------------------------------------
// GPTQ int4 dequant-GEMM for MI455X (gfx1250, wave32, WMMA + TDM).
//   y[M,N] = x[M,K] * W[N,K]^T,  W = (q4 - z4) * scale  (group size 128)
//   M = 8192, N = 12288, K = 4096
// Compute-bound (AI ~1470 FLOP/B): V_WMMA_F32_16X16X32_BF16 core,
// Tensor Data Mover (tensor_load_to_lds) moves A tiles (one instruction per
// workgroup per stage, DMA does the 2-D walk + LDS row padding), and
// packed-bf16 magic-number dequant (v_pk_fma_bf16) keeps VALU pressure low.
// ---------------------------------------------------------------------------

typedef __attribute__((ext_vector_type(16))) __bf16 v16bf;
typedef __attribute__((ext_vector_type(8)))  __bf16 v8bf;
typedef __attribute__((ext_vector_type(2)))  __bf16 v2bf;
typedef __attribute__((ext_vector_type(8)))  float  v8f;
typedef __attribute__((ext_vector_type(4)))  unsigned uint32x4;
typedef __attribute__((ext_vector_type(8)))  int      int32x8;
typedef __attribute__((ext_vector_type(4)))  int      int32x4;

#define K_DIM 4096
#define N_DIM 12288
#define GSZ   128
#define BK    64          // K per LDS stage (2 WMMA K=32 steps)
#define LDA   72          // padded bf16 row stride: 64 data + 8 pad (TDM pads 16B/row)
#define LDB   72

// Low 32 bits of a flat shared-memory address are the LDS byte address
// (flat->LDS mapping truncates to addr[31:0], CDNA5 ISA 10.2).
__device__ __forceinline__ unsigned lds_byte_addr(const void* p) {
  return (unsigned)(unsigned long long)p;
}

// ---------------------------------------------------------------------------
// Tensor Data Mover: load one 128-row x 64-col bf16 tile (row stride K_DIM)
// from global memory into LDS with 16B padding after every 128B row so the
// LDS layout matches LDA=72. One instruction moves the whole 16 KB tile;
// completion is tracked with TENSORcnt.
// D# layout per CDNA5 ISA. This toolchain exposes
// the 6-arg builtin: (g0, g1, g2, g3, g_extra, cpol).
// ---------------------------------------------------------------------------
__device__ __forceinline__ void tdm_load_a_tile(const void* gaddr, unsigned lds_addr,
                                                int m_rows) {
  unsigned long long ga = (unsigned long long)gaddr;
  uint32x4 g0;
  g0[0] = 1u;                                   // count=1, user descriptor
  g0[1] = lds_addr;                             // lds_addr (bytes)
  g0[2] = (unsigned)ga;                         // global_addr[31:0]
  g0[3] = (unsigned)(ga >> 32) | 0x80000000u;   // global_addr[56:32] | type=2

  int32x8 g1;
  // [17:16]=data_size(2B)=1, [20]=pad_enable, [24:22]=pad_interval(4 -> 32 DW
  // = 128B), [31:25]=pad_amount(3 -> 4 DW = 16B); workgroup_mask=0.
  g1[0] = 0x07110000;
  g1[1] = (int)((unsigned)K_DIM << 16);         // tensor_dim0[15:0] @ bits 63:48
  g1[2] = (int)((unsigned)m_rows << 16);        // tensor_dim1[15:0] @ bits 95:80
  g1[3] = (int)(64u << 16);                     // tile_dim0 = 64   @ bits 127:112
  g1[4] = 128;                                  // tile_dim1 = 128  @ bits 143:128
  g1[5] = K_DIM;                                // tensor_dim0_stride[31:0]
  g1[6] = 0;
  g1[7] = 0;

  int32x4 z4 = {0, 0, 0, 0};                    // 2-D tile: groups 2/3 unused
  int32x8 z8 = {0, 0, 0, 0, 0, 0, 0, 0};
  __builtin_amdgcn_tensor_load_to_lds(g0, g1, z4, z4, z8, 0);
}

// Magic-number dequant: bf16 pattern (0x4300 | q) == 128 + q exactly for
// q in [0,16). One packed FMA gives (q - z) * s for two weights at once:
//   w2 = (q+128)*s + (-(128+z)*s)
__device__ __forceinline__ unsigned dq2(unsigned qbits, v2bf s2, v2bf c2) {
  v2bf q = __builtin_bit_cast(v2bf, qbits | 0x43004300u);
  v2bf w = q * s2 + c2;              // contracts to v_pk_fma_bf16
  return __builtin_bit_cast(unsigned, w);
}

// ---------------------------------------------------------------------------
// Kernel 1: convert activations f32 -> bf16 once (amortized over 96 N-tiles).
// ---------------------------------------------------------------------------
__global__ void __launch_bounds__(256)
xcvt_f32_to_bf16(const float* __restrict__ x, __bf16* __restrict__ y, long long n) {
  long long i = ((long long)blockIdx.x * blockDim.x + threadIdx.x) * 8;
  if (i + 8 > n) return;
  float4 a = *(const float4*)(x + i);
  float4 b = *(const float4*)(x + i + 4);
  v8bf o;
  o[0] = (__bf16)a.x; o[1] = (__bf16)a.y; o[2] = (__bf16)a.z; o[3] = (__bf16)a.w;
  o[4] = (__bf16)b.x; o[5] = (__bf16)b.y; o[6] = (__bf16)b.z; o[7] = (__bf16)b.w;
  *(v8bf*)(y + i) = o;
}

// ---------------------------------------------------------------------------
// Kernel 2: tiled WMMA GEMM with TDM A-tiles and on-the-fly int4 dequant of B.
//   grid  = (N/128, M/128), block = 256 threads = 8 waves (wave32)
//   waves 4(M) x 2(N): each wave -> 32x64 output = 2x4 WMMA 16x16 tiles
// ---------------------------------------------------------------------------
__global__ void __launch_bounds__(256)
gptq_gemm_wmma(const __bf16*  __restrict__ xb,
               const uint8_t* __restrict__ qw,
               const float*   __restrict__ sc,
               const uint8_t* __restrict__ qz,
               float* __restrict__ out, int M) {
  __shared__ __bf16 At[2][128 * LDA];   // A tiles (TDM-filled), double buffered
  __shared__ __bf16 Bt[2][128 * LDB];   // dequantized B tiles, double buffered

  const int tid  = threadIdx.x;
  const int lane = tid & 31;
  const int half = (lane >> 4) & 1;     // lane group 0-15 / 16-31
  const int ln16 = lane & 15;
  const int wave = tid >> 5;
  const int wm   = wave >> 1;           // 0..3 -> M offset wm*32
  const int wn   = wave & 1;            // 0..1 -> N offset wn*64

  const int n0 = blockIdx.x * 128;
  const int m0 = blockIdx.y * 128;

  v8f acc[2][4];
#pragma unroll
  for (int mt = 0; mt < 2; ++mt)
#pragma unroll
    for (int nt = 0; nt < 4; ++nt)
#pragma unroll
      for (int r = 0; r < 8; ++r) acc[mt][nt][r] = 0.0f;

  // A-tile source pointer for stage kk (tile start inside the bf16 tensor).
  auto a_src = [&](int kk) -> const void* {
    return (const void*)(xb + (long long)m0 * K_DIM + kk * BK);
  };

  // Prologue: wave 0 kicks off the TDM for stage 0 (no prior readers).
  if (wave == 0) tdm_load_a_tile(a_src(0), lds_byte_addr(&At[0][0]), M);

  // Per-thread dequant assignment: 2 threads per B row, 16 packed bytes each.
  const int nl  = tid >> 1;             // local output row 0..127
  const int seg = tid & 1;              // 32-weight segment within BK
  const uint8_t* qbase = qw + (long long)(n0 + nl) * (K_DIM / 2) + seg * 16;
  const float*   sbase = sc + (long long)(n0 + nl) * (K_DIM / GSZ);
  const uint8_t* zbase = qz + (long long)(n0 + nl) * (K_DIM / (2 * GSZ));

  v2bf s2 = {};                         // {s, s} for current group
  v2bf c2 = {};                         // {-(128+z)*s, ...}

  const int NK = K_DIM / BK;            // 64 stages
  for (int kk = 0; kk < NK; ++kk) {
    const int b  = kk & 1;
    const int k0 = kk * BK;

    // ---- group constants change only every GSZ/BK = 2 stages ----
    if ((k0 % GSZ) == 0) {
      const int g = k0 / GSZ;
      float s = sbase[g];
      uint8_t zb = zbase[g >> 1];
      float z = (float)((g & 1) ? (zb >> 4) : (zb & 0xF));
      __bf16 sb = (__bf16)s;
      __bf16 cb = (__bf16)(-(128.0f + z) * s);
      s2[0] = sb; s2[1] = sb;
      c2[0] = cb; c2[1] = cb;
    }

    // ---- dequantize B(kk): 128 rows x 64 weights ----
    {
      const uint8_t* qrow = qbase + (k0 >> 1);
      uint4 qword = *(const uint4*)qrow;              // 32 packed int4
      uint32_t q4[4] = {qword.x, qword.y, qword.z, qword.w};
      __bf16* dst = &Bt[b][nl * LDB + seg * 32];
#pragma unroll
      for (int w = 0; w < 4; ++w) {                   // dword -> 8 weights
        uint32_t d = q4[w];
        uint4 r;
        // K pairs in order (lo,hi) per byte; hi nibble moved to bits 16-19.
        r.x = dq2((d & 0x0000000Fu) | ((d << 12) & 0x000F0000u), s2, c2);
        r.y = dq2(((d >> 8) & 0xFu) | ((d <<  4) & 0x000F0000u), s2, c2);
        r.z = dq2(((d >> 16) & 0xFu) | ((d >> 4) & 0x000F0000u), s2, c2);
        r.w = dq2(((d >> 24) & 0xFu) | ((d >> 12) & 0x000F0000u), s2, c2);
        *(uint4*)(dst + w * 8) = r;                   // ds_store_b128
      }
      if (kk + 1 < NK) __builtin_prefetch(qrow + 32, 0, 1);  // global_prefetch_b8
    }

    // Wave 0 owns the TDM: wait for A(kk) to land (TENSORcnt), then the
    // barrier publishes the LDS tile to all waves.
    if (wave == 0) __builtin_amdgcn_s_wait_tensorcnt(0);
    __syncthreads();

    // All waves are past compute(kk-1) -> safe to overwrite the other A
    // buffer; wave 0 launches the next TDM while everyone computes.
    if (kk + 1 < NK && wave == 0)
      tdm_load_a_tile(a_src(kk + 1), lds_byte_addr(&At[b ^ 1][0]), M);

    // ---- compute: two K=32 WMMA steps over this BK=64 stage ----
#pragma unroll
    for (int kt = 0; kt < 2; ++kt) {
      // A fragments: contiguous k order (same as B, not AMD's table order) -> lanes 0-15 hold K 0..15,
      // lanes 16-31 hold K 16..31: two contiguous 16B chunks per lane.
      v16bf afrag[2];
#pragma unroll
      for (int mt = 0; mt < 2; ++mt) {
        const __bf16* ap = &At[b][(wm * 32 + mt * 16 + ln16) * LDA + kt * 32];
        v8bf lo = *(const v8bf*)(ap + half * 16);
        v8bf hi = *(const v8bf*)(ap + half * 16 + 8);
#pragma unroll
        for (int i = 0; i < 8; ++i) { afrag[mt][i] = lo[i]; afrag[mt][8 + i] = hi[i]; }
      }
      // B fragments: contiguous k order, same as A -> lanes 0-15 hold K 0..15, lanes 16-31
      // hold K 16..31, N = lane&15: one contiguous 32B run per lane.
      v16bf bfrag[4];
#pragma unroll
      for (int nt = 0; nt < 4; ++nt) {
        const __bf16* bp =
            &Bt[b][(wn * 64 + nt * 16 + ln16) * LDB + kt * 32 + half * 16];
        v8bf lo = *(const v8bf*)(bp);
        v8bf hi = *(const v8bf*)(bp + 8);
#pragma unroll
        for (int i = 0; i < 8; ++i) { bfrag[nt][i] = lo[i]; bfrag[nt][8 + i] = hi[i]; }
      }
#pragma unroll
      for (int mt = 0; mt < 2; ++mt)
#pragma unroll
        for (int nt = 0; nt < 4; ++nt)
          acc[mt][nt] = __builtin_amdgcn_wmma_f32_16x16x32_bf16(
              false, afrag[mt], false, bfrag[nt],
              (short)0, acc[mt][nt], false, false);
    }
    // Single barrier per stage suffices with double buffering: stage kk+1
    // writes buffer b^1 whose last readers (stage kk-1) finished before
    // this stage's barrier.
  }

  // ---- epilogue: C/D 16x16 f32 layout: VGPR r -> M = r + 8*half, N = lane&15
#pragma unroll
  for (int mt = 0; mt < 2; ++mt)
#pragma unroll
    for (int nt = 0; nt < 4; ++nt) {
      int col = n0 + wn * 64 + nt * 16 + ln16;
#pragma unroll
      for (int r = 0; r < 8; ++r) {
        int row = m0 + wm * 32 + mt * 16 + r + 8 * half;
        out[(long long)row * N_DIM + col] = acc[mt][nt][r];
      }
    }
}

// ---------------------------------------------------------------------------
extern "C" void kernel_launch(void* const* d_in, const int* in_sizes, int n_in,
                              void* d_out, int out_size, void* d_ws, size_t ws_size,
                              hipStream_t stream) {
  const float*   x  = (const float*)d_in[0];
  const uint8_t* qw = (const uint8_t*)d_in[1];
  const float*   sc = (const float*)d_in[2];
  const uint8_t* qz = (const uint8_t*)d_in[3];
  // d_in[4] = group_size scalar (hardcoded 128 in kernels)
  float* out = (float*)d_out;

  long long nx = in_sizes[0];              // B*S*K = 33,554,432
  int M = (int)(nx / K_DIM);               // 8192
  __bf16* xb = (__bf16*)d_ws;              // bf16 activations (67 MB)

  long long nthreads = nx / 8;
  unsigned blocks = (unsigned)((nthreads + 255) / 256);
  xcvt_f32_to_bf16<<<blocks, 256, 0, stream>>>(x, xb, nx);

  dim3 grid(N_DIM / 128, M / 128);         // 96 x 64 workgroups
  gptq_gemm_wmma<<<grid, 256, 0, stream>>>(xb, qw, sc, qz, out, M);
}
